// APLoss_67276367725155
// MI455X (gfx1250) — compile-verified, matrix operand order checked by reading, not run on hardware
//
#include <hip/hip_runtime.h>
#include <hip/hip_bf16.h>

typedef __attribute__((ext_vector_type(16))) _Float16 v16h;
typedef __attribute__((ext_vector_type(8)))  float    v8f;
typedef __attribute__((ext_vector_type(4)))  float    v4f;

#define AP_N   262144
#define AP_FG  2048
#define AP_BG  (AP_N - AP_FG)      // 260096
#define AP_BGP 262144              // bg padded to multiple of 8192

// workspace layout (floats)
#define OFF_SORT 0
#define OFF_BGM  (AP_FG)                 // 2048
#define OFF_A    (AP_FG + AP_BGP)        // 264192
#define OFF_B    (OFF_A + AP_FG)         // 266240
#define OFF_THR  (OFF_B + AP_FG)         // 268288

// ---------------------------------------------------------------------------
// Kernel 1: bitonic-sort fg logits (2048) in LDS, write sorted + threshold,
// zero the a/b accumulators.
// ---------------------------------------------------------------------------
__global__ __launch_bounds__(1024)
void ap_prep_kernel(const float* __restrict__ logits,
                    float* __restrict__ sorted,
                    float* __restrict__ thr,
                    float* __restrict__ a,
                    float* __restrict__ b)
{
    __shared__ float s[AP_FG];
    const int t = threadIdx.x;                 // 1024 threads
    s[t]        = logits[t];
    s[t + 1024] = logits[t + 1024];
    a[t] = 0.0f; a[t + 1024] = 0.0f;
    b[t] = 0.0f; b[t + 1024] = 0.0f;
    __syncthreads();

    for (int k = 2; k <= AP_FG; k <<= 1) {
        for (int j = k >> 1; j > 0; j >>= 1) {
            #pragma unroll 1
            for (int base = 0; base < AP_FG; base += 1024) {
                const int i   = base + t;
                const int ixj = i ^ j;
                if (ixj > i) {
                    const float A = s[i];
                    const float B = s[ixj];
                    const bool  up = ((i & k) == 0);
                    // ascending: swap if A > B; descending: swap if A <= B
                    if ((A > B) == up) { s[i] = B; s[ixj] = A; }
                }
            }
            __syncthreads();
        }
    }

    sorted[t]        = s[t];
    sorted[t + 1024] = s[t + 1024];
    if (t == 0) thr[0] = s[0] - 1.0f;          // DELTA = 1.0
}

// ---------------------------------------------------------------------------
// Kernel 2: masked + padded copy of bg logits. Invalid / pad -> -1e30 so that
// clip((x - fv)*0.5 + 0.5, 0, 1) evaluates to exactly 0 (NaN-free).
// ---------------------------------------------------------------------------
__global__ __launch_bounds__(256)
void ap_mask_kernel(const float* __restrict__ logits,
                    const int*   __restrict__ targets,
                    const float* __restrict__ thr_p,
                    float* __restrict__ bgm)
{
    const int i = blockIdx.x * 256 + threadIdx.x;   // 0 .. AP_BGP-1
    const float thr = *thr_p;
    float v = -1.0e30f;
    if (i < AP_BG) {
        const float x = logits[AP_FG + i];
        if ((targets[AP_FG + i] == 0) && (x >= thr)) v = x;
    }
    bgm[i] = v;
}

// ---------------------------------------------------------------------------
// Kernel 3: WMMA accumulation.  out[f] += sum_x clip((x - fv_f)*0.5 + 0.5,0,1)
// Each wave: 16 fv columns (col = lane%16), x-slab of x_per_wave values.
// B[k][n] = t(x_k, fv_n) built in contiguous k order, k = 16*h + s
// (lane = 16*(k>=16) + n); A = ones => D[0][n] = full column sum (f32 accum).
// 4 independent tiles + accumulators per iteration so the post-WMMA hazard
// slots are filled with the next tile's fma/cvt work instead of v_nops.
// ---------------------------------------------------------------------------
__device__ __forceinline__ v16h ap_make_tile(const float* __restrict__ xp, float c)
{
    v16h bt;
    #pragma unroll
    for (int q = 0; q < 4; ++q) {
        const v4f x = *(const v4f*)(xp + 4 * q);
        #pragma unroll
        for (int e = 0; e < 4; ++e) {
            // lowers to v_fma_f32 ... clamp (fmed3 folded into output modifier)
            const float t = __builtin_amdgcn_fmed3f(fmaf(x[e], 0.5f, c), 0.0f, 1.0f);
            bt[4 * q + e] = (_Float16)t;
        }
    }
    return bt;
}

__global__ __launch_bounds__(256)
void ap_accum_kernel(const float* __restrict__ fvs,   // 2048 sorted fg values
                     const float* __restrict__ xs,    // x array (fg or masked bg)
                     float* __restrict__ out,         // a[] or b[] (2048)
                     int x_per_wave)                  // multiple of 128
{
    const int lane = threadIdx.x & 31;
    const int wave = threadIdx.x >> 5;                // 8 waves / block
    const int col  = lane & 15;                       // fv column
    const int hw   = lane >> 4;                       // half-wave -> K slab
    const int fv_base = blockIdx.x << 4;

    const float fv = fvs[fv_base + col];
    const float c  = 0.5f - 0.5f * fv;                // t = med3(x*0.5 + c, 0, 1)

    const float* xp = xs + (size_t)blockIdx.y * (size_t)(8 * x_per_wave)
                         + (size_t)wave * (size_t)x_per_wave
                         + (size_t)(hw * 16);

    v16h ones;
    #pragma unroll
    for (int i = 0; i < 16; ++i) ones[i] = (_Float16)1.0f;

    v8f acc0 = {};
    v8f acc1 = {};
    v8f acc2 = {};
    v8f acc3 = {};

    for (int it = 0; it < x_per_wave; it += 128) {
        // build all four B tiles first (independent register sets), then
        // issue the four WMMAs back-to-back on independent accumulators
        const v16h b0 = ap_make_tile(xp,      c);
        const v16h b1 = ap_make_tile(xp + 32, c);
        const v16h b2 = ap_make_tile(xp + 64, c);
        const v16h b3 = ap_make_tile(xp + 96, c);
        acc0 = __builtin_amdgcn_wmma_f32_16x16x32_f16(
                   false, ones, false, b0, (short)0, acc0, false, false);
        acc1 = __builtin_amdgcn_wmma_f32_16x16x32_f16(
                   false, ones, false, b1, (short)0, acc1, false, false);
        acc2 = __builtin_amdgcn_wmma_f32_16x16x32_f16(
                   false, ones, false, b2, (short)0, acc2, false, false);
        acc3 = __builtin_amdgcn_wmma_f32_16x16x32_f16(
                   false, ones, false, b3, (short)0, acc3, false, false);
        xp += 128;
    }

    const float sum = (acc0[0] + acc1[0]) + (acc2[0] + acc3[0]);
    if (lane < 16) atomicAdd(&out[fv_base + lane], sum);   // D[0][n], n = lane
}

// ---------------------------------------------------------------------------
// Kernel 4: cur = (a+0.5)/(a+0.5+b); inclusive prefix-max over sorted order; sum;
// metric = 1 - sum/FG.
// ---------------------------------------------------------------------------
__global__ __launch_bounds__(1024)
void ap_final_kernel(const float* __restrict__ a,
                     const float* __restrict__ b,
                     float* __restrict__ out)
{
    __shared__ float s0[AP_FG];
    __shared__ float s1[AP_FG];
    const int t = threadIdx.x;                 // 1024 threads

    for (int i = t; i < AP_FG; i += 1024) {
        const float av = a[i] + 0.5f;
        const float bv = b[i];
        s0[i] = av / (av + bv);
    }
    __syncthreads();

    float* src = s0;
    float* dst = s1;
    for (int off = 1; off < AP_FG; off <<= 1) {
        for (int i = t; i < AP_FG; i += 1024) {
            float v = src[i];
            if (i >= off) v = fmaxf(v, src[i - off]);
            dst[i] = v;
        }
        __syncthreads();
        float* tmp = src; src = dst; dst = tmp;
    }

    // src now holds the inclusive prefix-max (== prec_sorted); tree-sum it.
    dst[t] = src[t] + src[t + 1024];
    __syncthreads();
    for (int w = 512; w > 0; w >>= 1) {
        if (t < w) dst[t] += dst[t + w];
        __syncthreads();
    }
    if (t == 0) out[0] = 1.0f - dst[0] / (float)AP_FG;
}

// ---------------------------------------------------------------------------
extern "C" void kernel_launch(void* const* d_in, const int* in_sizes, int n_in,
                              void* d_out, int out_size, void* d_ws, size_t ws_size,
                              hipStream_t stream)
{
    (void)in_sizes; (void)n_in; (void)out_size; (void)ws_size;

    const float* logits  = (const float*)d_in[0];
    const int*   targets = (const int*)d_in[1];
    float* ws = (float*)d_ws;

    float* sorted = ws + OFF_SORT;
    float* bgm    = ws + OFF_BGM;
    float* a      = ws + OFF_A;
    float* b      = ws + OFF_B;
    float* thr    = ws + OFF_THR;

    // 1) sort fg, threshold, zero accumulators
    ap_prep_kernel<<<1, 1024, 0, stream>>>(logits, sorted, thr, a, b);

    // 2) masked + padded bg copy
    ap_mask_kernel<<<AP_BGP / 256, 256, 0, stream>>>(logits, targets, thr, bgm);

    // 3a) a[f] = sum over fg   : 8 waves * 256 x = 2048 per block row
    ap_accum_kernel<<<dim3(AP_FG / 16, 1), 256, 0, stream>>>(sorted, sorted, a, 256);

    // 3b) b[f] = sum over bg   : 8 waves * 1024 x = 8192 ; 32 chunks = 262144
    ap_accum_kernel<<<dim3(AP_FG / 16, 32), 256, 0, stream>>>(sorted, bgm, b, 1024);

    // 4) prefix-max scan + reduction -> metric
    ap_final_kernel<<<1, 1024, 0, stream>>>(a, b, (float*)d_out);
}
